// GCN_D_85950885527879
// MI455X (gfx1250) — matrix operand order repaired and checked by reading, not compiled, not run on hardware
//
#include <hip/hip_runtime.h>

// ---------------------------------------------------------------------------
// GCN on MI455X (gfx1250): three (adj*dis)@support aggregations done with
// v_wmma_f32_16x16x32_bf16 (HBM-bound at 23.3 TB/s: the 1.6 GB adj/dis stream
// dominates; bf16 WMMA keeps every pass off the fp32-matrix compute ceiling).
// adj/dis streamed once; product converted to bf16 on the fly and staged in
// LDS in the exact WMMA A-operand per-lane layout, double-buffered with a
// software-pipelined (peeled-tail) loop: one barrier per 128-j chunk, pure
// immediate-offset loads off auto-advancing cursors. Support matrices stored
// transposed bf16 so each lane's B operand is one contiguous 32B load (L2-hot).
// ---------------------------------------------------------------------------

typedef __attribute__((ext_vector_type(16))) __bf16 v16bf;
typedef __attribute__((ext_vector_type(8)))  float  v8f;

__device__ __forceinline__ unsigned short f2bf(float f) {
  // round-to-nearest-even fp32 -> bf16
  unsigned int u = __float_as_uint(f);
  unsigned int r = (u + 0x7fffu + ((u >> 16) & 1u)) >> 16;
  return (unsigned short)r;
}
__device__ __forceinline__ unsigned int pack2bf(float x, float y) {
  return (unsigned int)f2bf(x) | ((unsigned int)f2bf(y) << 16);
}

#define NNODES 8192

// ---------------------------------------------------------------------------
// Generic small dense layer: out[i,o] = act(sum_k in[i,k]*W[k,o] + b[o])
// act: 0=none 1=relu 2=leaky(0.2). tbf16: store bf16 TRANSPOSED [OUT][NNODES].
// ---------------------------------------------------------------------------
template <int IN, int OUT>
__global__ __launch_bounds__(256)
void lin_kernel(const float* __restrict__ in, const float* __restrict__ W,
                const float* __restrict__ bias, void* __restrict__ out,
                int act, int tbf16) {
  constexpr int ROWS = 256 / OUT;
  __shared__ float s_in[ROWS * IN];
  const int tid = threadIdx.x;
  const size_t row0 = (size_t)blockIdx.x * ROWS;

#pragma unroll
  for (int e = tid; e < ROWS * IN; e += 256)
    s_in[e] = in[row0 * IN + e];
  __syncthreads();

  const int rl = tid / OUT;
  const int o  = tid % OUT;
  float acc = bias ? bias[o] : 0.0f;
  const float* wp = W + o;
  const float* xi = s_in + rl * IN;
#pragma unroll
  for (int k = 0; k < IN; ++k)
    acc = fmaf(xi[k], wp[(size_t)k * OUT], acc);

  if (act == 1)      acc = fmaxf(acc, 0.0f);
  else if (act == 2) acc = (acc > 0.0f) ? acc : 0.2f * acc;

  const size_t grow = row0 + rl;
  if (tbf16)
    ((unsigned short*)out)[(size_t)o * NNODES + grow] = f2bf(acc);
  else
    ((float*)out)[grow * OUT + o] = acc;
}

// ---------------------------------------------------------------------------
// Aggregation: out[i,f] = act( sum_j bf16(adj[i,j]*dis[i,j]) * StT[f][j]
//                              + bias[f] + (res ? res[i,f] : 0) )
// StT: bf16 support transposed [K][NNODES].  K in {32,64,128} (compile-time).
// Block = 256 threads = 8 waves; FT=K/16 feature tiles x RT=8/FT row tiles.
// Software pipeline: stage chunk j+128 into LDS buffer p^1 while the WMMAs
// consume buffer p; peeled tail removes the guard from the hot loop.
// Staging: each thread owns (row m, 8 consecutive j): 2x b128 adj + 2x b128
// dis -> 4x v_pk_mul_f32 -> packed bf16 -> one 16B ds_store, landing directly
// in WMMA A-operand per-lane layout. Compute: preload 4 A (ds) + 4 B (global)
// operand vectors, then 4 back-to-back WMMAs on 2 accumulators.
// ---------------------------------------------------------------------------
template <int K>
__global__ __launch_bounds__(256)
void agg_kernel(const float* __restrict__ adj, const float* __restrict__ dis,
                const unsigned short* __restrict__ StT,
                const float* __restrict__ bias, const float* __restrict__ res,
                float* __restrict__ out, int act) {
  constexpr int N  = NNODES;
  constexpr int FT = K >> 4;      // feature tiles (8,4,2)
  constexpr int RT = 8 / FT;      // row tiles    (1,2,4)
  constexpr int ROWS = RT << 4;
  constexpr int BUF = RT * 4 * 512;          // halves per LDS buffer
  __shared__ unsigned short s_a[2 * BUF];    // double-buffered A tiles

  const int tid  = threadIdx.x;
  const int w    = tid >> 5;                 // wave id (wave32)
  const int lane = tid & 31;
  const int rt = w / FT;
  const int ft = w % FT;
  const int f0 = ft << 4;
  const int n   = lane & 15;
  const int hlf = lane >> 4;

  const size_t i0 = (size_t)blockIdx.x * ROWS;

  // staging role: 16 threads per row, 8 consecutive j each
  const int m   = tid >> 4;                  // row within 16-row tile
  const int j8  = (tid & 15) * 8;            // j offset within 128-chunk
  const int c4s = j8 >> 5;                   // subchunk this thread feeds
  const int ln  = m + (((j8 >> 4) & 1) << 4);      // target lane in A layout
  const int hh0 = ((j8 >> 3) & 1) << 3;            // target half base
  const int sidx = c4s * 512 + ln * 16 + hh0;      // half-index inside a tile

  // auto-advancing cursors (+128 elements per chunk); unrolled tile
  // displacements (t*16*N floats) become 24-bit immediate offsets.
  const float* ap = adj + (i0 + m) * (size_t)N + j8;
  const float* dp = dis + (i0 + m) * (size_t)N + j8;
  const unsigned short* bp = StT + (size_t)(f0 + n) * N + (hlf << 4);

  v8f c0 = {0.f, 0.f, 0.f, 0.f, 0.f, 0.f, 0.f, 0.f};
  v8f c1 = {0.f, 0.f, 0.f, 0.f, 0.f, 0.f, 0.f, 0.f};

  auto stage = [&](int p) {
#pragma unroll
    for (int t = 0; t < RT; ++t) {
      constexpr size_t TS = 16 * (size_t)N;  // row-tile displacement (floats)
      const float* at = ap + t * TS;
      const float* dt = dp + t * TS;
      __builtin_prefetch(at + 128, 0, 1);
      __builtin_prefetch(dt + 128, 0, 1);
      const float4 av0 = *(const float4*)(at);
      const float4 av1 = *(const float4*)(at + 4);
      const float4 dv0 = *(const float4*)(dt);
      const float4 dv1 = *(const float4*)(dt + 4);
      uint4 pk;
      pk.x = pack2bf(av0.x * dv0.x, av0.y * dv0.y);
      pk.y = pack2bf(av0.z * dv0.z, av0.w * dv0.w);
      pk.z = pack2bf(av1.x * dv1.x, av1.y * dv1.y);
      pk.w = pack2bf(av1.z * dv1.z, av1.w * dv1.w);
      *(uint4*)&s_a[p * BUF + t * 2048 + sidx] = pk;
    }
    ap += 128;
    dp += 128;
  };

  auto compute = [&](int p) {
    v16bf a0 = *(const v16bf*)(const void*)&s_a[p * BUF + (rt * 4 + 0) * 512 + lane * 16];
    v16bf a1 = *(const v16bf*)(const void*)&s_a[p * BUF + (rt * 4 + 1) * 512 + lane * 16];
    v16bf a2 = *(const v16bf*)(const void*)&s_a[p * BUF + (rt * 4 + 2) * 512 + lane * 16];
    v16bf a3 = *(const v16bf*)(const void*)&s_a[p * BUF + (rt * 4 + 3) * 512 + lane * 16];
    v16bf b0 = *(const v16bf*)(const void*)(bp + 0);
    v16bf b1 = *(const v16bf*)(const void*)(bp + 32);
    v16bf b2 = *(const v16bf*)(const void*)(bp + 64);
    v16bf b3 = *(const v16bf*)(const void*)(bp + 96);
    c0 = __builtin_amdgcn_wmma_f32_16x16x32_bf16(false, a0, false, b0, (short)0, c0, false, false);
    c1 = __builtin_amdgcn_wmma_f32_16x16x32_bf16(false, a1, false, b1, (short)0, c1, false, false);
    c0 = __builtin_amdgcn_wmma_f32_16x16x32_bf16(false, a2, false, b2, (short)0, c0, false, false);
    c1 = __builtin_amdgcn_wmma_f32_16x16x32_bf16(false, a3, false, b3, (short)0, c1, false, false);
    bp += 128;
  };

  // ---- software pipeline with peeled tail (no guard in the hot loop) ------
  stage(0);
  __syncthreads();
  int p = 0;
#pragma unroll 1
  for (int it = 0; it < N / 128 - 1; ++it) {
    stage(p ^ 1);        // overlap next HBM chunk with current WMMAs
    compute(p);
    __syncthreads();     // publish p^1, retire p
    p ^= 1;
  }
  compute(p);            // last chunk, nothing left to stage

  // ---- epilogue: bias + residual + activation, f32 row-major store --------
  const float bv = bias ? bias[f0 + n] : 0.0f;
#pragma unroll
  for (int r = 0; r < 8; ++r) {
    const size_t gi = i0 + (size_t)rt * 16 + r + hlf * 8;
    float v = c0[r] + c1[r] + bv;
    if (res) v += res[gi * K + f0 + n];
    if (act) v = fmaxf(v, 0.0f);
    out[gi * K + f0 + n] = v;
  }
}

// ---------------------------------------------------------------------------
extern "C" void kernel_launch(void* const* d_in, const int* in_sizes, int n_in,
                              void* d_out, int out_size, void* d_ws, size_t ws_size,
                              hipStream_t stream) {
  const float* x   = (const float*)d_in[0];
  const float* adj = (const float*)d_in[1];
  const float* dis = (const float*)d_in[2];
  const float* W1  = (const float*)d_in[3];  const float* b1  = (const float*)d_in[4];
  const float* Wg1 = (const float*)d_in[5];  const float* bg1 = (const float*)d_in[6];
  const float* Wm1 = (const float*)d_in[7];  const float* bm1 = (const float*)d_in[8];
  const float* Wm2 = (const float*)d_in[9];  const float* bm2 = (const float*)d_in[10];
  const float* Wm3 = (const float*)d_in[11]; const float* bm3 = (const float*)d_in[12];
  const float* Wm4 = (const float*)d_in[13]; const float* bm4 = (const float*)d_in[14];
  const float* Wgh = (const float*)d_in[15]; const float* bgh = (const float*)d_in[16];
  const float* W2  = (const float*)d_in[17]; const float* b2  = (const float*)d_in[18];
  const float* Wg2 = (const float*)d_in[19]; const float* bg2 = (const float*)d_in[20];
  const float* W3  = (const float*)d_in[21]; const float* b3  = (const float*)d_in[22];

  char* ws = (char*)d_ws;
  // three ping-pong regions, 4 MB each (12 MB total)
  char* A = ws;
  char* B = ws + (4u << 20);
  char* C = ws + (8u << 20);

#define LIN(IN_, OUT_, in_, W_, b_, out_, act_, t_)                          \
  lin_kernel<IN_, OUT_><<<NNODES / (256 / OUT_), 256, 0, stream>>>(          \
      (const float*)(in_), (W_), (b_), (void*)(out_), (act_), (t_))
#define AGG(K_, St_, b_, res_, out_, act_)                                   \
  agg_kernel<K_><<<NNODES / (16 * (8 / (K_ / 16))), 256, 0, stream>>>(       \
      adj, dis, (const unsigned short*)(St_), (b_), (const float*)(res_),    \
      (float*)(out_), (act_))

  // x1 = x@W1 + b1                                   -> A (f32 [8192,128])
  LIN(128, 128, x, W1, b1, A, 0, 0);
  // St1 = (x1@Wg1)^T bf16                            -> B [128][8192]
  LIN(128, 128, A, Wg1, nullptr, B, 0, 1);
  // g1 = (adj*dis)@support1 + bg1                    -> A (f32 [8192,128])
  AGG(128, B, bg1, nullptr, A, 0);
  // MLP chain
  LIN(128, 64, A, Wm1, bm1, C, 2, 0);   // lrelu -> C [8192,64]
  LIN(64, 128, C, Wm2, bm2, B, 2, 0);   // lrelu -> B [8192,128]
  LIN(128, 64, B, Wm3, bm3, C, 2, 0);   // lrelu -> C [8192,64]
  LIN(64, 64, C, Wm4, bm4, A, 1, 0);    // relu  -> A = x_res [8192,64]
  // St2 = (x_res@Wgh)^T bf16                         -> B [64][8192]
  LIN(64, 64, A, Wgh, nullptr, B, 0, 1);
  // h1 = relu((adj*dis)@support2 + bgh + x_res)      -> C [8192,64]
  AGG(64, B, bgh, A, C, 1);
  // t4 = h1@W2 + b2                                  -> B [8192,32]
  LIN(64, 32, C, W2, b2, B, 0, 0);
  // St3 = (t4@Wg2)^T bf16                            -> A [32][8192]
  LIN(32, 32, B, Wg2, nullptr, A, 0, 1);
  // g3 = (adj*dis)@support3 + bg2                    -> C [8192,32]
  AGG(32, A, bg2, nullptr, C, 0);
  // out = g3@W3 + b3                                 -> d_out (f32 [8192,16])
  LIN(32, 16, C, W3, b3, d_out, 0, 0);

#undef LIN
#undef AGG
}
